// TtLlamaQKV_optimized_80436147519617
// MI455X (gfx1250) — compile-verified, matrix operand order checked by reading, not run on hardware
//
#include <hip/hip_runtime.h>

// ---------------------------------------------------------------------------
// Fused Llama QKV projection for MI455X (gfx1250, wave32, WMMA).
//   M = SEQ = 2048, K = HIDDEN = 8192, N = 64*128 + 8*128 + 8*128 = 10240
// Strategy:
//   1. pack_a / pack_b: fp32 -> bf16, pre-swizzled into the exact per-lane
//      VGPR fragment layout of V_WMMA_F32_16X16X32_BF16 (ISA 7.12.2).
//      bf16 A (32MB) + bf16 W (160MB) = 192MB -> resident in the 192MB L2.
//      Source fp32 reads are non-temporal (read-once, don't pollute L2).
//   2. qkv_gemm: 128x256 block tile, 8 waves (2x4), 64x64 wave tile
//      (4x4 WMMA accumulators -> 2.0 WMMA per fragment fetched, 32 FLOP/B
//      from cache), 16 v_wmma_f32_16x16x32_bf16 per K=32 step.
//      Epilogue streams (non-temporal) straight into the (head, seq, dim)
//      permuted output so the 84MB of results never evict A/B from L2.
// Workspace: A_pack 33,554,432 B + B_pack 167,772,160 B  (~201.4 MB).
// ---------------------------------------------------------------------------

typedef __attribute__((ext_vector_type(16))) __bf16 v16bf;
typedef __attribute__((ext_vector_type(8)))  float  v8f;
typedef __attribute__((ext_vector_type(4)))  float  v4f;

#define SEQ     2048
#define HIDDEN  8192
#define NTOT    10240          // 8192 q + 1024 k + 1024 v
#define KTILES  (HIDDEN / 32)  // 256
#define MTILES  (SEQ / 16)     // 128
#define NTILES  (NTOT / 16)    // 640

// ---- fragment packing -----------------------------------------------------
// Fragment f(t, kt) for a 16x32 (A) or 32x16 (B) bf16 tile:
//   lane L (0..31): row/col = t*16 + (L&15), koff = (L>>4)*8
//   elems 0..7  = K = kt*32 + koff + j
//   elems 8..15 = K = kt*32 + 16 + koff + j
// stored contiguously: frag_base = ((t*KTILES + kt)*32 + L) * 16 bf16.

__global__ __launch_bounds__(256) void pack_a(const float* __restrict__ x,
                                              __bf16* __restrict__ ap) {
    int tid  = blockIdx.x * 256 + threadIdx.x;     // = (mt*KTILES + kt)*32 + lane
    int lane = tid & 31;
    int kt   = (tid >> 5) & (KTILES - 1);
    int mt   = tid >> 13;
    int row   = mt * 16 + (lane & 15);
    int kbase = kt * 32 + ((lane >> 4) << 3);

    const v4f* src = (const v4f*)(x + (size_t)row * HIDDEN + kbase);
    v4f f0 = __builtin_nontemporal_load(src);
    v4f f1 = __builtin_nontemporal_load(src + 1);
    v4f f2 = __builtin_nontemporal_load(src + 4);
    v4f f3 = __builtin_nontemporal_load(src + 5);

    v16bf v;
#pragma unroll
    for (int j = 0; j < 4; ++j) {
        v[j]      = (__bf16)f0[j];
        v[4 + j]  = (__bf16)f1[j];
        v[8 + j]  = (__bf16)f2[j];
        v[12 + j] = (__bf16)f3[j];
    }

    *(v16bf*)(ap + (size_t)tid * 16) = v;   // regular store: want this in L2
}

__global__ __launch_bounds__(256) void pack_b(const float* __restrict__ wq,
                                              const float* __restrict__ wk,
                                              const float* __restrict__ wv,
                                              __bf16* __restrict__ bp) {
    int tid  = blockIdx.x * 256 + threadIdx.x;     // = (nt*KTILES + kt)*32 + lane
    int lane = tid & 31;
    int kt   = (tid >> 5) & (KTILES - 1);
    int nt   = tid >> 13;
    int n     = nt * 16 + (lane & 15);             // fused column
    int kbase = kt * 32 + ((lane >> 4) << 3);

    const float* w; int col, ldw;
    if (n < 8192)       { w = wq; col = n;        ldw = 8192; }
    else if (n < 9216)  { w = wk; col = n - 8192; ldw = 1024; }
    else                { w = wv; col = n - 9216; ldw = 1024; }

    const float* c0 = w + (size_t)kbase * ldw + col;
    v16bf v;
#pragma unroll
    for (int j = 0; j < 8; ++j)
        v[j] = (__bf16)__builtin_nontemporal_load(c0 + (size_t)j * ldw);
    const float* c1 = c0 + (size_t)16 * ldw;
#pragma unroll
    for (int j = 0; j < 8; ++j)
        v[8 + j] = (__bf16)__builtin_nontemporal_load(c1 + (size_t)j * ldw);

    *(v16bf*)(bp + (size_t)tid * 16) = v;   // regular store: want this in L2
}

// ---- GEMM -----------------------------------------------------------------
// 8 waves arranged 2 (M) x 4 (N); each wave owns a 64x64 tile:
// 4 A-tiles x 4 B-tiles -> 16 WMMAs per K=32 step on 8 fetched fragments.
__global__ __launch_bounds__(256) void qkv_gemm(const __bf16* __restrict__ Ap,
                                                const __bf16* __restrict__ Bp,
                                                float* __restrict__ out) {
    const int lane = threadIdx.x & 31;
    const int wave = threadIdx.x >> 5;
    const int wr   = wave >> 2;        // 0..1
    const int wc   = wave & 3;         // 0..3

    const int mt0 = blockIdx.y * 8  + wr * 4;   // four 16-row A tiles
    const int nt0 = blockIdx.x * 16 + wc * 4;   // four 16-col B tiles

    const v16bf* Av = (const v16bf*)Ap;
    const v16bf* Bv = (const v16bf*)Bp;

    size_t aIdx[4], bIdx[4];
#pragma unroll
    for (int i = 0; i < 4; ++i) {
        aIdx[i] = (size_t)(mt0 + i) * (KTILES * 32) + lane;
        bIdx[i] = (size_t)(nt0 + i) * (KTILES * 32) + lane;
    }

    v8f acc[4][4] = {};

    for (int kt = 0; kt < KTILES; ++kt) {
        const size_t ks = (size_t)kt * 32;
        v16bf a[4], b[4];
#pragma unroll
        for (int i = 0; i < 4; ++i) a[i] = Av[aIdx[i] + ks];
#pragma unroll
        for (int j = 0; j < 4; ++j) b[j] = Bv[bIdx[j] + ks];

#pragma unroll
        for (int i = 0; i < 4; ++i)
#pragma unroll
            for (int j = 0; j < 4; ++j)
                acc[i][j] = __builtin_amdgcn_wmma_f32_16x16x32_bf16(
                    false, a[i], false, b[j], (short)0, acc[i][j], false, false);
    }

    // Epilogue: C/D layout (ISA 7.12.2): VGPR r, lane L -> element
    //   (M = r + 8*(L>>4), N = L&15) of the 16x16 tile.
    // Output permutation: out[(n>>7)*SEQ*128 + s*128 + (n&127)].
    // Non-temporal: keep the 84MB of results out of L2 (A/B live there).
    const int mBase = mt0 * 16 + ((lane >> 4) << 3);
#pragma unroll
    for (int i = 0; i < 4; ++i) {
#pragma unroll
        for (int j = 0; j < 4; ++j) {
            int n    = (nt0 + j) * 16 + (lane & 15);
            int head = n >> 7;
            int d    = n & 127;
            float* ob = out + (size_t)head * SEQ * 128 + d;
            int s0 = mBase + i * 16;
#pragma unroll
            for (int r = 0; r < 8; ++r)
                __builtin_nontemporal_store(acc[i][j][r],
                                            &ob[(size_t)(s0 + r) * 128]);
        }
    }
}

// ---- host glue ------------------------------------------------------------
extern "C" void kernel_launch(void* const* d_in, const int* in_sizes, int n_in,
                              void* d_out, int out_size, void* d_ws, size_t ws_size,
                              hipStream_t stream) {
    const float* x  = (const float*)d_in[0];
    const float* wq = (const float*)d_in[1];
    const float* wk = (const float*)d_in[2];
    const float* wv = (const float*)d_in[3];
    float* out = (float*)d_out;

    __bf16* Ap = (__bf16*)d_ws;                                     // 32 MB
    __bf16* Bp = (__bf16*)((char*)d_ws + (size_t)SEQ * HIDDEN * 2); // 160 MB

    // fp32 -> bf16 fragment-swizzle passes (streaming, ~30 us class)
    pack_a<<<(MTILES * KTILES * 32) / 256, 256, 0, stream>>>(x, Ap);
    pack_b<<<(NTILES * KTILES * 32) / 256, 256, 0, stream>>>(wq, wk, wv, Bp);

    // BF16 WMMA GEMM, 128x256 block tiles: grid 40 x 16
    dim3 grid(NTOT / 256, SEQ / 128);
    qkv_gemm<<<grid, 256, 0, stream>>>(Ap, Bp, out);
}
